// CrossAttentionScorer_67894843015729
// MI455X (gfx1250) — matrix operand order repaired and checked by reading, not compiled, not run on hardware
//
#include <hip/hip_runtime.h>
#include <hip/hip_bf16.h>

// ---------------------------------------------------------------------------
// CrossAttentionScorer, fused for MI455X (gfx1250, wave32, WMMA).
// seq_len == 1 on both sides => softmax over one element == 1.0 exactly,
// so attention output == v projection; dead q/k path skipped (bit-exact).
//
// Pipeline (per 32-row block held in LDS end-to-end):
//   LN(candidate) -> v = x@wv.T+bv -> a = v@wo.T+bo -> h = relu(a@w1.T+b1)
//   -> out = h@w2.T+b2
// Weights are pre-packed to bf16 WMMA-B-fragment tiles in d_ws. GEMM hot
// loop (round-2 shape, the one this backend compiles cleanly): 1 LDS v16bf
// load (A, shared) + 2 global v16bf loads (B pair) -> 2 independent
// v_wmma_f32_16x16x32_bf16 chains. No fp32->bf16 VALU work in the loop.
// ---------------------------------------------------------------------------

typedef __attribute__((ext_vector_type(16))) __bf16 v16bf;
typedef __attribute__((ext_vector_type(8)))  float  v8f;

#define DIM     1024
#define FFD     2048
#define NROWS   32768
#define ROWS    32        // rows per workgroup (2 M-tiles of 16)
#define WAVES   8
#define THREADS 256

// float -> bf16, round to nearest even (bit ops only)
static __device__ __forceinline__ __bf16 f2bf(float f) {
    unsigned u = __builtin_bit_cast(unsigned, f);
    u += 0x7fffu + ((u >> 16) & 1u);
    unsigned short s = (unsigned short)(u >> 16);
    return __builtin_bit_cast(__bf16, s);
}
static __device__ __forceinline__ float bf2f(__bf16 b) {
    unsigned u = ((unsigned)__builtin_bit_cast(unsigned short, b)) << 16;
    return __builtin_bit_cast(float, u);
}

// Swap bits 3 and 4 of an index: permutes each 32-block of the K dimension
// into WMMA-A fragment order [0..7,16..23 | 8..15,24..31]. Involution.
static __device__ __forceinline__ int swz(int i) {
    return (i & ~24) | ((i & 8) << 1) | ((i & 16) >> 1);
}

// ---------------------------------------------------------------------------
// Prep: pack fp32 torch-layout weight W[N][K] into bf16 B-fragment tiles.
// Tile (nt, kt) covers cols nt*16..+15, K kt*32..+31. Stored as 32 lanes x
// 16 contiguous bf16: lane l -> col nt*16+(l&15), K kt*32 + swz((l<16?0:16) + j).
// ---------------------------------------------------------------------------
extern "C" __global__ void __launch_bounds__(256)
pack_weights_bf16(const float* __restrict__ W, __bf16* __restrict__ out,
                  int N, int K) {
    const int ktiles = K / 32;
    const int total  = (N / 16) * ktiles * 32;     // one thread per lane-slot
    int idx = blockIdx.x * blockDim.x + threadIdx.x;
    if (idx >= total) return;
    const int lane = idx & 31;
    const int tile = idx >> 5;                     // nt*ktiles + kt
    const int kt   = tile % ktiles;
    const int nt   = tile / ktiles;
    const int n    = nt * 16 + (lane & 15);
    const int kb   = (lane < 16) ? 0 : 16;
    const float* src = W + (size_t)n * K + kt * 32;
    __bf16* dst = out + (size_t)tile * 512 + lane * 16;
    #pragma unroll
    for (int j = 0; j < 16; ++j) dst[j] = f2bf(src[swz(kb + j)]);
}

// ---------------------------------------------------------------------------
// LayerNorm 32 rows (4 per wave): fp32 global -> bf16 LDS (K-swizzled)
// ---------------------------------------------------------------------------
static __device__ void layernorm_rows(const float* __restrict__ x,
                                      const float* __restrict__ w,
                                      const float* __restrict__ b,
                                      __bf16* __restrict__ out,
                                      int wave, int lane) {
    for (int r = 0; r < ROWS / WAVES; ++r) {
        const int row = wave * (ROWS / WAVES) + r;
        const float4* xr4 = (const float4*)(x + (size_t)row * DIM);
        float s = 0.f, s2 = 0.f;
        for (int i = lane; i < DIM / 4; i += 32) {
            float4 v = xr4[i];
            s  += v.x + v.y + v.z + v.w;
            s2 += v.x * v.x + v.y * v.y + v.z * v.z + v.w * v.w;
        }
        for (int off = 16; off > 0; off >>= 1) {
            s  += __shfl_down(s,  off, 32);
            s2 += __shfl_down(s2, off, 32);
        }
        s  = __shfl(s,  0, 32);
        s2 = __shfl(s2, 0, 32);
        const float mu  = s * (1.0f / DIM);
        const float var = s2 * (1.0f / DIM) - mu * mu;
        const float inv = rsqrtf(var + 1e-5f);
        for (int i = lane; i < DIM / 4; i += 32) {
            float4 v = xr4[i];
            const int i0 = i * 4;                  // 4 consecutive, same 8-block
            const int p  = row * DIM + swz(i0);    // swz keeps bits0-2 -> contiguous
            out[p + 0] = f2bf((v.x - mu) * inv * w[i0 + 0] + b[i0 + 0]);
            out[p + 1] = f2bf((v.y - mu) * inv * w[i0 + 1] + b[i0 + 1]);
            out[p + 2] = f2bf((v.z - mu) * inv * w[i0 + 2] + b[i0 + 2]);
            out[p + 3] = f2bf((v.w - mu) * inv * w[i0 + 3] + b[i0 + 3]);
        }
    }
}

// ---------------------------------------------------------------------------
// GEMM stage: actOut[m,n] = act( sum_k actIn[m,k]*W[n,k] + bias[n] )
// actIn: LDS bf16 [ROWS][K], K-swizzled. Wsw: global bf16 fragment tiles.
// Each wave: adjacent N-tile pairs (two acc chains, shared A fragment).
// ---------------------------------------------------------------------------
static __device__ void wave_gemm_stage(const __bf16* __restrict__ actIn, int ldi,
                                       const __bf16* __restrict__ Wsw,
                                       const float*  __restrict__ bias,
                                       __bf16* __restrict__ actOut, int ldo,
                                       int N, int K, bool relu,
                                       int wave, int lane) {
    const int ntiles = N / 16;
    const int ktiles = K / 32;
    const int npairs = (ROWS / 16) * ntiles / 2;

    const int halfhi = (lane >= 16);
    const int lane15 = lane & 15;

    for (int p = wave; p < npairs; p += WAVES) {
        const int t0  = 2 * p;
        const int m0  = (t0 / ntiles) * 16;
        const int nt0 = t0 % ntiles;
        const int nt1 = nt0 + 1;
        const int arow = m0 + lane15;

        v8f acc0 = {0.f, 0.f, 0.f, 0.f, 0.f, 0.f, 0.f, 0.f};
        v8f acc1 = {0.f, 0.f, 0.f, 0.f, 0.f, 0.f, 0.f, 0.f};

        const __bf16* ap  = actIn + (size_t)arow * ldi + (halfhi ? 16 : 0);
        const __bf16* b0p = Wsw + ((size_t)nt0 * ktiles) * 512 + lane * 16;
        const __bf16* b1p = Wsw + ((size_t)nt1 * ktiles) * 512 + lane * 16;

        for (int kt = 0; kt < ktiles; ++kt) {
            v16bf a  = *(const v16bf*)(ap + kt * 32);       // 32B LDS load
            v16bf b0 = *(const v16bf*)(b0p + kt * 512);     // 32B global load
            v16bf b1 = *(const v16bf*)(b1p + kt * 512);
            acc0 = __builtin_amdgcn_wmma_f32_16x16x32_bf16(
                false, a, false, b0, (short)0, acc0, false, false);
            acc1 = __builtin_amdgcn_wmma_f32_16x16x32_bf16(
                false, a, false, b1, (short)0, acc1, false, false);
        }

        // epilogue: C/D layout (VGPR r: M = mbase+r, N = lane15 of the tile)
        const int ncol0 = nt0 * 16 + lane15;
        const int ncol1 = nt1 * 16 + lane15;
        const float bv0 = bias[ncol0];
        const float bv1 = bias[ncol1];
        const int sc0 = swz(ncol0);                // store at swizzled K-pos
        const int sc1 = swz(ncol1);
        const int mbase = m0 + (halfhi ? 8 : 0);
        #pragma unroll
        for (int r = 0; r < 8; ++r) {
            float v0 = acc0[r] + bv0;
            float v1 = acc1[r] + bv1;
            if (relu) { v0 = v0 > 0.f ? v0 : 0.f; v1 = v1 > 0.f ? v1 : 0.f; }
            actOut[(size_t)(mbase + r) * ldo + sc0] = f2bf(v0);
            actOut[(size_t)(mbase + r) * ldo + sc1] = f2bf(v1);
        }
    }
}

// Final projection: out[row] = sum_f h[row][f]*w2[f] + b2 (h is K-swizzled)
static __device__ void final_dot(const __bf16* __restrict__ h,
                                 const float* __restrict__ w2, float b2,
                                 float* __restrict__ out, int rowbase,
                                 int wave, int lane) {
    for (int r = 0; r < ROWS / WAVES; ++r) {
        const int row = wave * (ROWS / WAVES) + r;
        const __bf16* hr = h + (size_t)row * FFD;
        float s = 0.f;
        for (int j = lane; j < FFD; j += 32)
            s += bf2f(hr[j]) * w2[swz(j)];
        for (int off = 16; off > 0; off >>= 1)
            s += __shfl_down(s, off, 32);
        if (lane == 0) out[rowbase + row] = s + b2;
    }
}

extern "C" __global__ void __launch_bounds__(THREADS)
cross_attn_scorer_fused(const float*  __restrict__ cand,
                        const float*  __restrict__ ln_kv_w,
                        const float*  __restrict__ ln_kv_b,
                        const __bf16* __restrict__ wv16, const float* __restrict__ bv,
                        const __bf16* __restrict__ wo16, const float* __restrict__ bo,
                        const __bf16* __restrict__ w116, const float* __restrict__ b1,
                        const float*  __restrict__ w2,   const float* __restrict__ b2,
                        float* __restrict__ out) {
    extern __shared__ char smem[];
    __bf16* bufA = (__bf16*)smem;                      // [ROWS][DIM]  64 KB
    __bf16* bufH = (__bf16*)(smem + ROWS * DIM * 2);   // [ROWS][FFD] 128 KB

    const int lane = threadIdx.x & 31;
    const int wave = threadIdx.x >> 5;
    const int rowbase = blockIdx.x * ROWS;

    layernorm_rows(cand + (size_t)rowbase * DIM, ln_kv_w, ln_kv_b, bufA, wave, lane);
    __syncthreads();

    // v = kv @ wv.T + bv   (softmax over a single key == 1 -> attn out == v)
    wave_gemm_stage(bufA, DIM, wv16, bv, bufH, DIM, DIM, DIM, false, wave, lane);
    __syncthreads();

    // a = v @ wo.T + bo
    wave_gemm_stage(bufH, DIM, wo16, bo, bufA, DIM, DIM, DIM, false, wave, lane);
    __syncthreads();

    // h = relu(a @ w1.T + b1)
    wave_gemm_stage(bufA, DIM, w116, b1, bufH, FFD, FFD, DIM, true, wave, lane);
    __syncthreads();

    // out = h @ w2.T + b2
    final_dot(bufH, w2, b2[0], out, rowbase, wave, lane);
}

extern "C" void kernel_launch(void* const* d_in, const int* in_sizes, int n_in,
                              void* d_out, int out_size, void* d_ws, size_t ws_size,
                              hipStream_t stream) {
    // setup_inputs() order:
    //  0 context_feature, 1 candidate_feature,
    //  2 ln_q_w, 3 ln_q_b, 4 ln_kv_w, 5 ln_kv_b,
    //  6 wq, 7 wk, 8 wv, 9 bq, 10 bk, 11 bv, 12 wo, 13 bo,
    // 14 w1, 15 b1, 16 w2, 17 b2
    const float* cand    = (const float*)d_in[1];
    const float* ln_kv_w = (const float*)d_in[4];
    const float* ln_kv_b = (const float*)d_in[5];
    const float* wv      = (const float*)d_in[8];
    const float* bv      = (const float*)d_in[11];
    const float* wo      = (const float*)d_in[12];
    const float* bo      = (const float*)d_in[13];
    const float* w1      = (const float*)d_in[14];
    const float* b1      = (const float*)d_in[15];
    const float* w2      = (const float*)d_in[16];
    const float* b2      = (const float*)d_in[17];
    float* out           = (float*)d_out;

    // Workspace: bf16 fragment-packed weights (8 MB total)
    __bf16* wv16 = (__bf16*)d_ws;                       // 1024*1024
    __bf16* wo16 = wv16 + (size_t)DIM * DIM;            // 1024*1024
    __bf16* w116 = wo16 + (size_t)DIM * DIM;            // 2048*1024

    // Pack weights -> bf16 fragment tiles (every call; deterministic)
    {
        int tv = (DIM / 16) * (DIM / 32) * 32;          // 65536 threads
        int tf = (FFD / 16) * (DIM / 32) * 32;          // 131072 threads
        pack_weights_bf16<<<dim3((tv + 255) / 256), dim3(256), 0, stream>>>(wv, wv16, DIM, DIM);
        pack_weights_bf16<<<dim3((tv + 255) / 256), dim3(256), 0, stream>>>(wo, wo16, DIM, DIM);
        pack_weights_bf16<<<dim3((tf + 255) / 256), dim3(256), 0, stream>>>(w1, w116, FFD, DIM);
    }

    const size_t lds_bytes = (size_t)ROWS * DIM * 2 + (size_t)ROWS * FFD * 2; // 192 KB
    dim3 grid(NROWS / ROWS);   // 1024 workgroups
    dim3 block(THREADS);       // 8 wave32
    cross_attn_scorer_fused<<<grid, block, lds_bytes, stream>>>(
        cand, ln_kv_w, ln_kv_b, wv16, bv, wo16, bo, w116, b1, w2, b2, out);
}
